// MultiheadFlexAttention_78847009620254
// MI455X (gfx1250) — matrix operand order repaired and checked by reading, not compiled, not run on hardware
//
#include <hip/hip_runtime.h>

// ---------------------------------------------------------------------------
// MultiheadFlexAttention for MI455X (gfx1250, wave32, WMMA + TDM)
//   q = X Wq^T + bq ; k,v likewise  -> f16 [B,H,S,D]
//   flash attention (f16 WMMA, f32 accum, online softmax,
//                    double-buffered KV staging with TDM overlap)
//   out = attn Wo^T + bo            -> f32 [B,S,E]
// ---------------------------------------------------------------------------

typedef __attribute__((ext_vector_type(16))) _Float16 v16h;
typedef __attribute__((ext_vector_type(8)))  float    v8f;
typedef __attribute__((ext_vector_type(8)))  _Float16 h8;
typedef __attribute__((ext_vector_type(4)))  _Float16 h4;

#define EMBED 1024
#define HEADS 16
#define HDIM  64
#define BATCH 2
#define SEQ   2048
#define MTOT  (BATCH * SEQ)   // 4096 rows for the projection GEMMs
#define NKV   (SEQ / 64)      // 32 KV tiles per head

#if __has_builtin(__builtin_amdgcn_tensor_load_to_lds) && \
    __has_builtin(__builtin_amdgcn_s_wait_tensorcnt)
#define USE_TDM 1
#warning "CDNA5: TDM tensor_load_to_lds path ENABLED"
#else
#define USE_TDM 0
#warning "CDNA5: TDM builtins NOT available - manual staging fallback"
#endif

// ---------------------------------------------------------------------------
// WMMA helpers (V_WMMA_F32_16X16X32_F16, wave32)
// ---------------------------------------------------------------------------
__device__ __forceinline__ v8f wmma_f16(v16h a, v16h b, v8f c) {
  // 8 args: (neg_a, A, neg_b, B, c_mod, C, reuse_a, reuse_b)
  return __builtin_amdgcn_wmma_f32_16x16x32_f16(false, a, false, b,
                                                (short)0, c, false, false);
}

// A-matrix 16x32 (MxK), 16-bit (ISA 7.12.2):
//   lane L: M = L%16 ; k0 = (L/16)*8
//   elems 0..7  -> K = k0 + i        elems 8..15 -> K = k0 + 16 + (i-8)
__device__ __forceinline__ v16h load_a_frag(const _Float16* base, int ld, int lane) {
  const int m  = lane & 15;
  const int k0 = (lane >> 4) << 3;
  const _Float16* p0 = base + m * ld + k0;
  const _Float16* p1 = p0 + 16;
  v16h a;
#pragma unroll
  for (int i = 0; i < 8; ++i) { a[i] = p0[i]; a[8 + i] = p1[i]; }
  return a;
}

// B-matrix 32x16 (KxN), 16-bit: lane L holds col N=L%16, k0=(L/16)*8;
//   elems 0..7 -> K = k0 + i        elems 8..15 -> K = k0 + 16 + (i-8)
__device__ __forceinline__ v16h load_b_frag(const _Float16* base, int ld, int lane) {
  const _Float16* p = base + ((lane >> 4) << 3) * ld + (lane & 15);
  v16h b;
#pragma unroll
  for (int i = 0; i < 8; ++i) { b[i] = p[i * ld]; b[8 + i] = p[(16 + i) * ld]; }
  return b;
}

// C/D 16x16 f32: VGPR j, lane L -> row M = j + 8*(L/16), col N = L%16.

#if USE_TDM
// ---------------------------------------------------------------------------
// Tensor Data Mover: load a 2D tile of f16 (tile_d0 x tile_d1, row-major,
// row stride `stride_elems`) from global into LDS at byte offset `lds_off`.
// LDS padding: after every 32 DWORDs (one 64-half row) insert 4 DWORDs
// (8 halfs) -> LDS row pitch 72 halfs, matching the shared-tile layout.
// D# layout per CDNA5 ISA §8.3 (group 0) and §8.4 (group 1).
// ---------------------------------------------------------------------------
typedef __attribute__((ext_vector_type(4))) unsigned int tdm_u32x4;
typedef __attribute__((ext_vector_type(8))) int          tdm_i32x8;
typedef __attribute__((ext_vector_type(4))) int          tdm_i32x4;

__device__ __forceinline__ void tdm_load_tile_f16(unsigned int lds_off,
                                                  const _Float16* gptr,
                                                  unsigned int tile_d0,
                                                  unsigned int tile_d1,
                                                  unsigned int stride_elems,
                                                  bool pad_rows)
{
  const unsigned long long ga = (unsigned long long)gptr;
  tdm_u32x4 g0;
  g0[0] = 1u;                                   // count=1, user descriptor
  g0[1] = lds_off;                              // lds_addr (bytes)
  g0[2] = (unsigned int)ga;                     // global_addr[31:0]
  g0[3] = (unsigned int)((ga >> 32) & 0x1FFFFFFu) | (2u << 30); // addr[56:32]|type=2

  // group1: [15:0] wg_mask=0, [17:16] data_size=1 (2B), [20] pad_enable,
  // [24:22] pad_interval (code 4 -> 32 DWORDs), [31:25] pad_amount (code 3 -> 4 DWORDs)
  unsigned int ctl = (1u << 16);
  if (pad_rows) ctl |= (1u << 20) | (4u << 22) | (3u << 25);
  tdm_i32x8 g1;
  g1[0] = (int)ctl;
  g1[1] = (int)((tile_d0 & 0xFFFFu) << 16);     // tensor_dim0[15:0] @ bits 63:48
  g1[2] = (int)(((tile_d0 >> 16) & 0xFFFFu) |   // tensor_dim0[31:16]
                ((tile_d1 & 0xFFFFu) << 16));   // tensor_dim1[15:0]
  g1[3] = (int)(((tile_d1 >> 16) & 0xFFFFu) |   // tensor_dim1[31:16]
                ((tile_d0 & 0xFFFFu) << 16));   // tile_dim0
  g1[4] = (int)(tile_d1 & 0xFFFFu);             // tile_dim1 ; tile_dim2 = 0
  g1[5] = (int)stride_elems;                    // tensor_dim0_stride[31:0]
  g1[6] = 0;                                    // stride[47:32] / dim1_stride lo
  g1[7] = 0;

  tdm_i32x4 z4 = {0, 0, 0, 0};
#if defined(__clang_major__) && (__clang_major__ >= 23)
  tdm_i32x8 z8 = {0, 0, 0, 0, 0, 0, 0, 0};
  __builtin_amdgcn_tensor_load_to_lds(g0, g1, z4, z4, z8, 0);
#else
  __builtin_amdgcn_tensor_load_to_lds(g0, g1, z4, z4, 0);
#endif
}
#endif  // USE_TDM

// ---------------------------------------------------------------------------
// Kernel 1: QKV projection.  C[m,n] = sum_k X[m,k]*W[n,k] + b[n]
// X: f32 [MTOT, EMBED] ; W: f32 [EMBED, EMBED] ; out: f16 in [B,H,S,D].
// Block = 128 thr (4 waves), tile 64(M) x 64(N), K-step 32.
// ---------------------------------------------------------------------------
__global__ __launch_bounds__(128)
void qkv_proj_kernel(const float* __restrict__ X, const float* __restrict__ W,
                     const float* __restrict__ bias, _Float16* __restrict__ out_bhsd)
{
  __shared__ _Float16 As[64][40];   // [m][k], pitch 40 halfs
  __shared__ _Float16 Wt[32][72];   // [k][n], W transposed, pitch 72 halfs

  const int t = threadIdx.x;
  const int lane = t & 31, wave = t >> 5;
  const int m0 = blockIdx.x * 64;
  const int n0 = blockIdx.y * 64;

  v8f acc[4] = {};   // 4 N-chunks of 16

  for (int kt = 0; kt < EMBED; kt += 32) {
    // stage: 64x32 f32 -> f16, vectorized float4 loads (8 vecs per row)
#pragma unroll
    for (int i = 0; i < 4; ++i) {
      const int v = t + i * 128;              // 512 float4 vectors
      const int r = v >> 3, c4 = (v & 7) * 4;
      const float4 xv = *(const float4*)&X[(size_t)(m0 + r) * EMBED + kt + c4];
      h4 xh = {(_Float16)xv.x, (_Float16)xv.y, (_Float16)xv.z, (_Float16)xv.w};
      *(h4*)&As[r][c4] = xh;                  // 8B-aligned ds_store_b64
      const float4 wv = *(const float4*)&W[(size_t)(n0 + r) * EMBED + kt + c4];
      Wt[c4 + 0][r] = (_Float16)wv.x;         // transposed scatter
      Wt[c4 + 1][r] = (_Float16)wv.y;
      Wt[c4 + 2][r] = (_Float16)wv.z;
      Wt[c4 + 3][r] = (_Float16)wv.w;
    }
    __syncthreads();

    const v16h a = load_a_frag(&As[wave * 16][0], 40, lane);
#pragma unroll
    for (int c = 0; c < 4; ++c) {
      const v16h b = load_b_frag(&Wt[0][c * 16], 72, lane);
      acc[c] = wmma_f16(a, b, acc[c]);
    }
    __syncthreads();
  }

  // Epilogue: bias + scatter f16 into [B,H,S,D]
  const int hi = lane >> 4, col = lane & 15;
#pragma unroll
  for (int c = 0; c < 4; ++c) {
    const int n = n0 + c * 16 + col;
    const float bv = bias[n];
    const int h = n >> 6, d = n & 63;
#pragma unroll
    for (int j = 0; j < 8; ++j) {
      const int m = m0 + wave * 16 + j + 8 * hi;
      const int b = m >> 11, s = m & (SEQ - 1);   // SEQ = 2048
      out_bhsd[(((size_t)b * HEADS + h) * SEQ + s) * HDIM + d] =
          (_Float16)(acc[c][j] + bv);
    }
  }
}

// ---------------------------------------------------------------------------
// Kernel 2: flash attention.  Q,K,V f16 [B,H,S,D] -> O f16 [B,S,H,D].
// Block = 128 thr (4 waves) handles one (b, h, 64-row q tile).
// KV tiles double-buffered: tile i+1 staged (TDM V-load issued async +
// manual K transpose) while tile i is consumed by the WMMAs; one barrier
// per iteration; s_wait_tensorcnt deferred to just before the barrier.
// ---------------------------------------------------------------------------
__global__ __launch_bounds__(128)
void flash_attn_kernel(const _Float16* __restrict__ Q, const _Float16* __restrict__ K,
                       const _Float16* __restrict__ V, _Float16* __restrict__ O)
{
  __shared__ _Float16 Qs[64][72];         // [q][d]
  __shared__ _Float16 Kt[2][64][72];      // [d][kv]  (transposed on stage)
  __shared__ _Float16 Vs[2][64][72];      // [kv][d]
  __shared__ _Float16 Ps[4][16][72];      // per-wave probabilities [q][kv]

  const int t = threadIdx.x, lane = t & 31, wave = t >> 5;
  const int q0 = blockIdx.x * 64;
  const int h  = blockIdx.y;
  const int b  = blockIdx.z;
  const size_t head_base = ((size_t)b * HEADS + h) * SEQ;

  // ---- stage one KV tile into buffer `buf`
  auto stage_tile = [&](int buf, int kv0) {
#if USE_TDM
    if (wave == 0) {
      tdm_load_tile_f16((unsigned int)(uintptr_t)&Vs[buf][0][0],
                        &V[(head_base + kv0) * HDIM], 64, 64, HDIM, true);
    }
#endif
#pragma unroll
    for (int i = 0; i < 4; ++i) {
      const int v = t + i * 128;            // 512 vecs of 8 halfs
      const int r = v >> 3, c8 = (v & 7) * 8;
      const h8 kv8 = *(const h8*)&K[(head_base + kv0 + r) * HDIM + c8];
#pragma unroll
      for (int j = 0; j < 8; ++j) Kt[buf][c8 + j][r] = kv8[j];  // transpose
#if !USE_TDM
      *(h8*)&Vs[buf][r][c8] = *(const h8*)&V[(head_base + kv0 + r) * HDIM + c8];
#endif
    }
  };

  // ---- prologue: stage Q tile + KV tile 0
#if USE_TDM
  if (wave == 0) {
    tdm_load_tile_f16((unsigned int)(uintptr_t)&Qs[0][0],
                      &Q[(head_base + q0) * HDIM], 64, 64, HDIM, true);
  }
#else
#pragma unroll
  for (int i = 0; i < 4; ++i) {
    const int v = t + i * 128;
    const int r = v >> 3, c8 = (v & 7) * 8;
    *(h8*)&Qs[r][c8] = *(const h8*)&Q[(head_base + q0 + r) * HDIM + c8];
  }
#endif
  stage_tile(0, 0);
#if USE_TDM
  if (wave == 0) __builtin_amdgcn_s_wait_tensorcnt(0);
#endif
  __syncthreads();

  const int hi = lane >> 4, col = lane & 15;
  float m_run[8], l_run[8];
#pragma unroll
  for (int j = 0; j < 8; ++j) { m_run[j] = -3.0e38f; l_run[j] = 0.0f; }
  v8f acc_o[4] = {};   // O accum over 4 d-chunks (f32)

  for (int it = 0; it < NKV; ++it) {
    const int buf = it & 1;

    // ---- kick off staging of the NEXT tile before computing this one
    if (it + 1 < NKV) {
      stage_tile(buf ^ 1, (it + 1) * 64);
      if (it + 2 < NKV) {                 // warm L2 two tiles ahead
        __builtin_prefetch(&K[(head_base + (it + 2) * 64 + (t >> 1)) * HDIM], 0, 0);
        __builtin_prefetch(&V[(head_base + (it + 2) * 64 + (t >> 1)) * HDIM], 0, 0);
      }
    }

    // ---- S = (Q K^T) * 1/sqrt(D) : 16x64 per wave, K-dim = 64 (2 steps)
    v8f s_acc[4] = {};
    const v16h a0 = load_a_frag(&Qs[wave * 16][0],  72, lane);
    const v16h a1 = load_a_frag(&Qs[wave * 16][32], 72, lane);
#pragma unroll
    for (int c = 0; c < 4; ++c) {
      const v16h b0 = load_b_frag(&Kt[buf][0][c * 16],  72, lane);
      s_acc[c] = wmma_f16(a0, b0, s_acc[c]);
      const v16h b1 = load_b_frag(&Kt[buf][32][c * 16], 72, lane);
      s_acc[c] = wmma_f16(a1, b1, s_acc[c]);
    }
#pragma unroll
    for (int c = 0; c < 4; ++c)
#pragma unroll
      for (int j = 0; j < 8; ++j) s_acc[c][j] *= 0.125f;  // 1/sqrt(64)

    // ---- online softmax over the 64 kv columns of this tile
#pragma unroll
    for (int j = 0; j < 8; ++j) {
      float mx = -3.0e38f;
#pragma unroll
      for (int c = 0; c < 4; ++c) mx = fmaxf(mx, s_acc[c][j]);
      for (int off = 1; off < 16; off <<= 1)        // reduce 16-lane row group
        mx = fmaxf(mx, __shfl_xor(mx, off, 32));
      const float mnew  = fmaxf(m_run[j], mx);
      const float alpha = __expf(m_run[j] - mnew);
      m_run[j] = mnew;
      float ls = 0.0f;
#pragma unroll
      for (int c = 0; c < 4; ++c) {
        const float p = __expf(s_acc[c][j] - mnew);
        s_acc[c][j] = p;
        ls += p;
      }
      for (int off = 1; off < 16; off <<= 1)
        ls += __shfl_xor(ls, off, 32);
      l_run[j] = l_run[j] * alpha + ls;
#pragma unroll
      for (int c = 0; c < 4; ++c) acc_o[c][j] *= alpha;
    }

    // ---- stage P (f16) through per-wave LDS to re-enter as an A fragment
#pragma unroll
    for (int c = 0; c < 4; ++c)
#pragma unroll
      for (int j = 0; j < 8; ++j)
        Ps[wave][j + 8 * hi][c * 16 + col] = (_Float16)s_acc[c][j];
    // (same-wave ds write->read; compiler inserts s_wait_dscnt)

    // ---- O += P @ V : K-dim = 64 kv (2 steps), 4 d-chunks
    const v16h pa0 = load_a_frag(&Ps[wave][0][0],  72, lane);
    const v16h pa1 = load_a_frag(&Ps[wave][0][32], 72, lane);
#pragma unroll
    for (int c = 0; c < 4; ++c) {
      const v16h vb0 = load_b_frag(&Vs[buf][0][c * 16],  72, lane);
      acc_o[c] = wmma_f16(pa0, vb0, acc_o[c]);
      const v16h vb1 = load_b_frag(&Vs[buf][32][c * 16], 72, lane);
      acc_o[c] = wmma_f16(pa1, vb1, acc_o[c]);
    }

    // ---- publish next buffer / retire this one
    if (it + 1 < NKV) {
#if USE_TDM
      if (wave == 0) __builtin_amdgcn_s_wait_tensorcnt(0);
#endif
      __syncthreads();
    }
  }

  // Epilogue: normalize and write f16 O in [B,S,H,D] (= [B,S,E] concat)
#pragma unroll
  for (int j = 0; j < 8; ++j) l_run[j] = 1.0f / l_run[j];
#pragma unroll
  for (int c = 0; c < 4; ++c) {
    const int d = c * 16 + col;
#pragma unroll
    for (int j = 0; j < 8; ++j) {
      const int s = q0 + wave * 16 + j + 8 * hi;
      O[(((size_t)b * SEQ + s) * HEADS + h) * HDIM + d] =
          (_Float16)(acc_o[c][j] * l_run[j]);
    }
  }
}

// ---------------------------------------------------------------------------
// Kernel 3: output projection.  out[m,n] = sum_k A[m,k]*Wo[n,k] + bo[n]
// A: f16 [MTOT, EMBED] (attn output) ; out: f32 [MTOT, EMBED] = d_out.
// ---------------------------------------------------------------------------
__global__ __launch_bounds__(128)
void out_proj_kernel(const _Float16* __restrict__ A, const float* __restrict__ W,
                     const float* __restrict__ bias, float* __restrict__ out)
{
  __shared__ _Float16 As[64][40];
  __shared__ _Float16 Wt[32][72];

  const int t = threadIdx.x;
  const int lane = t & 31, wave = t >> 5;
  const int m0 = blockIdx.x * 64;
  const int n0 = blockIdx.y * 64;

  v8f acc[4] = {};

  for (int kt = 0; kt < EMBED; kt += 32) {
#pragma unroll
    for (int i = 0; i < 2; ++i) {         // A: 256 vecs of 8 halfs (b128)
      const int v = t + i * 128;
      const int r = v >> 2, c8 = (v & 3) * 8;
      *(h8*)&As[r][c8] = *(const h8*)&A[(size_t)(m0 + r) * EMBED + kt + c8];
    }
#pragma unroll
    for (int i = 0; i < 4; ++i) {         // W: 512 float4 vecs, transposed
      const int v = t + i * 128;
      const int r = v >> 3, c4 = (v & 7) * 4;
      const float4 wv = *(const float4*)&W[(size_t)(n0 + r) * EMBED + kt + c4];
      Wt[c4 + 0][r] = (_Float16)wv.x;
      Wt[c4 + 1][r] = (_Float16)wv.y;
      Wt[c4 + 2][r] = (_Float16)wv.z;
      Wt[c4 + 3][r] = (_Float16)wv.w;
    }
    __syncthreads();

    const v16h a = load_a_frag(&As[wave * 16][0], 40, lane);
#pragma unroll
    for (int c = 0; c < 4; ++c) {
      const v16h b = load_b_frag(&Wt[0][c * 16], 72, lane);
      acc[c] = wmma_f16(a, b, acc[c]);
    }
    __syncthreads();
  }

  const int hi = lane >> 4, col = lane & 15;
#pragma unroll
  for (int c = 0; c < 4; ++c) {
    const int n = n0 + c * 16 + col;
    const float bv = bias[n];
#pragma unroll
    for (int j = 0; j < 8; ++j) {
      const int m = m0 + wave * 16 + j + 8 * hi;
      out[(size_t)m * EMBED + n] = acc[c][j] + bv;
    }
  }
}

// ---------------------------------------------------------------------------
// Launch
// ---------------------------------------------------------------------------
extern "C" void kernel_launch(void* const* d_in, const int* in_sizes, int n_in,
                              void* d_out, int out_size, void* d_ws, size_t ws_size,
                              hipStream_t stream)
{
  const float* query = (const float*)d_in[0];
  const float* key   = (const float*)d_in[1];
  const float* value = (const float*)d_in[2];
  const float* Wq    = (const float*)d_in[3];
  const float* bq    = (const float*)d_in[4];
  const float* Wk    = (const float*)d_in[5];
  const float* bk    = (const float*)d_in[6];
  const float* Wv    = (const float*)d_in[7];
  const float* bv    = (const float*)d_in[8];
  const float* Wo    = (const float*)d_in[9];
  const float* bo    = (const float*)d_in[10];
  float* out = (float*)d_out;

  // Workspace: 4 x (4096*1024) f16 buffers = 32 MB
  _Float16* qws = (_Float16*)d_ws;                    // [B,H,S,D]
  _Float16* kws = qws + (size_t)MTOT * EMBED;         // [B,H,S,D]
  _Float16* vws = kws + (size_t)MTOT * EMBED;         // [B,H,S,D]
  _Float16* aws = vws + (size_t)MTOT * EMBED;         // [B,S,H,D]

  dim3 gProj(MTOT / 64, EMBED / 64);                  // (64, 16)
  qkv_proj_kernel<<<gProj, 128, 0, stream>>>(query, Wq, bq, qws);
  qkv_proj_kernel<<<gProj, 128, 0, stream>>>(key,   Wk, bk, kws);
  qkv_proj_kernel<<<gProj, 128, 0, stream>>>(value, Wv, bv, vws);

  dim3 gAttn(SEQ / 64, HEADS, BATCH);                 // (32, 16, 2)
  flash_attn_kernel<<<gAttn, 128, 0, stream>>>(qws, kws, vws, aws);

  out_proj_kernel<<<gProj, 128, 0, stream>>>(aws, Wo, bo, out);
}
